// NNCF_RNN_893353197874
// MI455X (gfx1250) — matrix operand order repaired and checked by reading, not compiled, not run on hardware
//
#include <hip/hip_runtime.h>

// ---------------------------------------------------------------------------
// Bidirectional 2-layer LSTM (T=256, B=64, I=512, H=512) for MI455X (gfx1250)
//
// Recurrence-latency-bound. bf16 WMMA (v_wmma_f32_16x16x32_bf16) for the
// per-step (64 x 2048 x K) gate GEMM, fp32 accumulate. Weights packed
// [Wih | Whh] bf16 (L2-resident, ~20MB total << 192MB L2).
// State placement tuned for register pressure (round-1 disasm showed scratch
// spills): h double-buffered bf16 in LDS, c fp32 in LDS (padded leading dims
// for bank spread), bias staged in LDS, only ONE tile-group's accumulators
// (4 x v8f) live in VGPRs at a time. One workgroup (16 wave32) per direction.
// ---------------------------------------------------------------------------

typedef __bf16 bf16_t;
typedef __attribute__((ext_vector_type(8)))  __bf16 v8bf;
typedef __attribute__((ext_vector_type(16))) __bf16 v16bf;
typedef __attribute__((ext_vector_type(8)))  float  v8f;

static constexpr int T_ = 256;
static constexpr int B_ = 64;
static constexpr int I_ = 512;
static constexpr int H_ = 512;
static constexpr int HLD = H_ + 8;   // bf16 h leading dim: row stride 1040B -> banks spread
static constexpr int CLD = H_ + 4;   // f32  c leading dim: rows 8 apart land 32 banks apart

// ---------------------------------------------------------------------------
// Fragment loaders matching CDNA5 ISA 7.12.2 VGPR layouts (wave32).
// A (16x32 bf16, MxK), row-major source with leading dim ld:
//   lanes 0-15:  M=lane,    K = {k0+0..7,  k0+16..23}
//   lanes 16-31: M=lane-16, K = {k0+8..15, k0+24..31}
// ---------------------------------------------------------------------------
__device__ __forceinline__ v16bf load_frag_a(const bf16_t* base, int ld, int lane) {
    const int m    = lane & 15;
    const int ksel = (lane & 16) ? 8 : 0;
    const bf16_t* p = base + (size_t)m * ld + ksel;
    v8bf lo = *(const v8bf*)(p);
    v8bf hi = *(const v8bf*)(p + 16);
    return __builtin_shufflevector(lo, hi, 0,1,2,3,4,5,6,7,8,9,10,11,12,13,14,15);
}

// B (32x16 bf16, KxN): column n of B = row n of row-major W (N x K).
// lanes 0-15: N=lane, K={k0+0..7, k0+16..23}; lanes 16-31: K={k0+8..15, k0+24..31}.
__device__ __forceinline__ v16bf load_frag_b(const bf16_t* base, int ld, int lane) {
    const int n    = lane & 15;
    const int ksel = (lane & 16) ? 8 : 0;
    const bf16_t* p = base + (size_t)n * ld + ksel;
    v8bf lo = *(const v8bf*)(p);
    v8bf hi = *(const v8bf*)(p + 16);
    return __builtin_shufflevector(lo, hi, 0,1,2,3,4,5,6,7,8,9,10,11,12,13,14,15);
}

__device__ __forceinline__ float sigm(float x) {
    return 1.0f / (1.0f + __expf(-x));
}
__device__ __forceinline__ float fast_tanh(float x) {
    float e = __expf(-2.0f * fabsf(x));
    float t = (1.0f - e) / (1.0f + e);
    return copysignf(t, x);
}

// ---------------------------------------------------------------------------
// Prep kernels
// ---------------------------------------------------------------------------
__global__ void cvt_f32_to_bf16(const float* __restrict__ src,
                                bf16_t* __restrict__ dst, long n) {
    long i = (long)blockIdx.x * blockDim.x + threadIdx.x;
    long stride = (long)gridDim.x * blockDim.x;
    for (; i < n; i += stride) dst[i] = (bf16_t)src[i];
}

__global__ void pack_weights(const float* __restrict__ wih,
                             const float* __restrict__ whh,
                             const float* __restrict__ bih,
                             const float* __restrict__ bhh,
                             bf16_t* __restrict__ wcat,
                             float*  __restrict__ bias,
                             int din) {
    const int  ldw   = din + H_;
    const long total = (long)4 * H_ * ldw;
    long i = (long)blockIdx.x * blockDim.x + threadIdx.x;
    long stride = (long)gridDim.x * blockDim.x;
    for (; i < total; i += stride) {
        int n = (int)(i / ldw), k = (int)(i % ldw);
        float v = (k < din) ? wih[(size_t)n * din + k]
                            : whh[(size_t)n * H_ + (k - din)];
        wcat[i] = (bf16_t)v;
    }
    for (long n = (long)blockIdx.x * blockDim.x + threadIdx.x; n < 4 * H_;
         n += stride)
        bias[n] = bih[n] + bhh[n];
}

// ---------------------------------------------------------------------------
// Persistent per-direction LSTM layer kernel.
//   gridDim.x = 2 (block 0 = forward, block 1 = reverse), blockDim = 512.
// Wave w owns M-tile (w&3) and 8 J-tiles ((w>>2)*8 + g). The four gate tiles
// (i,f,g,o) for one (m,j) accumulate together -> cell update fully local.
// ---------------------------------------------------------------------------
template <int DIN, bool OUT_F32>
__launch_bounds__(512, 1)
__global__ void lstm_layer_kernel(const bf16_t* __restrict__ xin,       // (T,B,DIN)
                                  const bf16_t* __restrict__ wcat_base, // 2 x (4H, DIN+H)
                                  const float*  __restrict__ bias_base, // 2 x 4H
                                  const float*  __restrict__ h0_base,   // 2 x (B,H)
                                  const float*  __restrict__ c0_base,   // 2 x (B,H)
                                  const int*    __restrict__ lengths,   // (B)
                                  bf16_t* __restrict__ out_bf,          // (T,B,2H) bf16
                                  float*  __restrict__ out_f) {         // (T,B,2H) f32
    constexpr int LDW = DIN + H_;

    extern __shared__ char smem_raw[];
    bf16_t* hbuf0  = (bf16_t*)smem_raw;                       // 64 x HLD bf16
    bf16_t* hbuf1  = hbuf0 + B_ * HLD;                        // 64 x HLD bf16
    float*  cbuf   = (float*)(hbuf1 + B_ * HLD);              // 64 x CLD f32
    float*  bias_s = cbuf + B_ * CLD;                         // 4H f32

    const int dir  = blockIdx.x;               // 0 = fwd, 1 = reverse
    const int tid  = threadIdx.x;
    const int lane = tid & 31;
    const int wave = tid >> 5;

    const bf16_t* Wc   = wcat_base + (size_t)dir * (size_t)4 * H_ * LDW;
    const float*  bias = bias_base + (size_t)dir * 4 * H_;
    const float*  h0   = h0_base + (size_t)dir * B_ * H_;
    const float*  c0   = c0_base + (size_t)dir * B_ * H_;

    const int m0    = (wave & 3) * 16;             // M-tile origin (batch rows)
    const int jbase = (wave >> 2) * 8;             // first owned J tile
    const int rlo   = m0 + ((lane & 16) ? 8 : 0);  // C/D row half per lane group
    const int nloc  = lane & 15;                   // C/D column within tile

    // --- init LDS state ---------------------------------------------------
    for (int i = tid; i < B_ * H_; i += blockDim.x) {
        int r = i >> 9, c = i & (H_ - 1);
        hbuf0[r * HLD + c] = (bf16_t)h0[i];
        cbuf[r * CLD + c]  = c0[i];
    }
    for (int i = tid; i < 4 * H_; i += blockDim.x) bias_s[i] = bias[i];

    // per-lane sequence lengths for the 8 C/D rows this lane owns (fixed)
    int len_r[8];
#pragma unroll
    for (int r = 0; r < 8; ++r) len_r[r] = lengths[rlo + r];

    __syncthreads();

    // --- time loop --------------------------------------------------------
    for (int step = 0; step < T_; ++step) {
        const int t = (dir == 1) ? (T_ - 1 - step) : step;
        const bf16_t* xt   = xin + (size_t)t * B_ * DIN;
        const bf16_t* hsrc = (step & 1) ? hbuf1 : hbuf0;
        bf16_t*       hdst = (step & 1) ? hbuf0 : hbuf1;

#pragma unroll 1
        for (int g = 0; g < 8; ++g) {
            const int j0 = (jbase + g) * 16;

            // accumulators for the four gates, bias-initialized from LDS
            v8f acc[4];
#pragma unroll
            for (int gt = 0; gt < 4; ++gt) {
                const float bv = bias_s[gt * H_ + j0 + nloc];
#pragma unroll
                for (int r = 0; r < 8; ++r) acc[gt][r] = bv;
            }

            // phase 1: x_t @ Wih^T  (A fragments from global bf16 x)
#pragma unroll 2
            for (int k0 = 0; k0 < DIN; k0 += 32) {
                v16bf a = load_frag_a(xt + (size_t)m0 * DIN + k0, DIN, lane);
#pragma unroll
                for (int gt = 0; gt < 4; ++gt) {
                    v16bf b = load_frag_b(
                        Wc + (size_t)(gt * H_ + j0) * LDW + k0, LDW, lane);
                    acc[gt] = __builtin_amdgcn_wmma_f32_16x16x32_bf16(
                        false, a, false, b, (short)0, acc[gt], false, false);
                }
            }
            // phase 2: h @ Whh^T  (A fragments from LDS h, double-buffered)
#pragma unroll 2
            for (int k0 = 0; k0 < H_; k0 += 32) {
                v16bf a = load_frag_a(hsrc + m0 * HLD + k0, HLD, lane);
#pragma unroll
                for (int gt = 0; gt < 4; ++gt) {
                    v16bf b = load_frag_b(
                        Wc + (size_t)(gt * H_ + j0) * LDW + DIN + k0, LDW, lane);
                    acc[gt] = __builtin_amdgcn_wmma_f32_16x16x32_bf16(
                        false, a, false, b, (short)0, acc[gt], false, false);
                }
            }

            // cell update: i,f,g,o -> c,h ; c lives in LDS, masked by length
            const int col = j0 + nloc;
#pragma unroll
            for (int r = 0; r < 8; ++r) {
                const float iv = sigm(acc[0][r]);
                const float fv = sigm(acc[1][r]);
                const float gv = fast_tanh(acc[2][r]);
                const float ov = sigm(acc[3][r]);
                const int   row  = rlo + r;
                const float cold = cbuf[row * CLD + col];
                const float cn = fv * cold + iv * gv;
                const float hn = ov * fast_tanh(cn);
                const bool  mk = (t < len_r[r]);
                cbuf[row * CLD + col] = mk ? cn : cold;
                const float hold = (float)hsrc[row * HLD + col];
                hdst[row * HLD + col] = (bf16_t)(mk ? hn : hold);
                const float oval = mk ? hn : 0.0f;
                const size_t oidx =
                    ((size_t)t * B_ + row) * (2 * H_) + (size_t)dir * H_ + col;
                if (OUT_F32) out_f[oidx] = oval;
                else         out_bf[oidx] = (bf16_t)oval;
            }
        }
        __syncthreads();   // h double-buffer + c visibility for next step
    }
}

// ---------------------------------------------------------------------------
// Host side
// ---------------------------------------------------------------------------
extern "C" void kernel_launch(void* const* d_in, const int* in_sizes, int n_in,
                              void* d_out, int out_size, void* d_ws,
                              size_t ws_size, hipStream_t stream) {
    (void)in_sizes; (void)n_in; (void)out_size; (void)ws_size;

    const float* x       = (const float*)d_in[0];
    const int*   lengths = (const int*)  d_in[1];
    const float* h0      = (const float*)d_in[2];   // (4,B,H)
    const float* c0      = (const float*)d_in[3];   // (4,B,H)
    const float* w_ih[4] = {(const float*)d_in[4],  (const float*)d_in[8],
                            (const float*)d_in[12], (const float*)d_in[16]};
    const float* w_hh[4] = {(const float*)d_in[5],  (const float*)d_in[9],
                            (const float*)d_in[13], (const float*)d_in[17]};
    const float* b_ih[4] = {(const float*)d_in[6],  (const float*)d_in[10],
                            (const float*)d_in[14], (const float*)d_in[18]};
    const float* b_hh[4] = {(const float*)d_in[7],  (const float*)d_in[11],
                            (const float*)d_in[15], (const float*)d_in[19]};

    // workspace carve-up (256B aligned)
    char* ws = (char*)d_ws;
    size_t off = 0;
    auto carve = [&](size_t bytes) -> void* {
        void* p = ws + off;
        off = (off + bytes + 255) & ~(size_t)255;
        return p;
    };
    const size_t LDW0 = (size_t)(I_ + H_);        // 1024
    const size_t LDW1 = (size_t)(2 * H_ + H_);    // 1536
    bf16_t* xbf   = (bf16_t*)carve((size_t)T_ * B_ * I_ * 2);
    bf16_t* h1bf  = (bf16_t*)carve((size_t)T_ * B_ * 2 * H_ * 2);
    bf16_t* wcat0 = (bf16_t*)carve((size_t)2 * 4 * H_ * LDW0 * 2);
    bf16_t* wcat1 = (bf16_t*)carve((size_t)2 * 4 * H_ * LDW1 * 2);
    float*  bias0 = (float*)carve((size_t)2 * 4 * H_ * 4);
    float*  bias1 = (float*)carve((size_t)2 * 4 * H_ * 4);

    // 1) x -> bf16
    cvt_f32_to_bf16<<<2048, 256, 0, stream>>>(x, xbf, (long)T_ * B_ * I_);

    // 2) pack [Wih|Whh] bf16 + combined bias, per cell
    pack_weights<<<1024, 256, 0, stream>>>(w_ih[0], w_hh[0], b_ih[0], b_hh[0],
                                           wcat0, bias0, I_);
    pack_weights<<<1024, 256, 0, stream>>>(w_ih[1], w_hh[1], b_ih[1], b_hh[1],
                                           wcat0 + (size_t)4 * H_ * LDW0,
                                           bias0 + 4 * H_, I_);
    pack_weights<<<1024, 256, 0, stream>>>(w_ih[2], w_hh[2], b_ih[2], b_hh[2],
                                           wcat1, bias1, 2 * H_);
    pack_weights<<<1024, 256, 0, stream>>>(w_ih[3], w_hh[3], b_ih[3], b_hh[3],
                                           wcat1 + (size_t)4 * H_ * LDW1,
                                           bias1 + 4 * H_, 2 * H_);

    // LDS: 2 x h (bf16, HLD) + c (f32, CLD) + bias (f32)
    const size_t shmem = (size_t)2 * B_ * HLD * sizeof(bf16_t) +
                         (size_t)B_ * CLD * sizeof(float) +
                         (size_t)4 * H_ * sizeof(float);

    // 3) layer 0 (fwd + bwd concurrently), bf16 output feeds layer 1
    lstm_layer_kernel<I_, false><<<2, 512, shmem, stream>>>(
        xbf, wcat0, bias0, h0, c0, lengths, h1bf, nullptr);

    // 4) layer 1 (fwd + bwd), fp32 output straight to d_out
    lstm_layer_kernel<2 * H_, true><<<2, 512, shmem, stream>>>(
        h1bf, wcat1, bias1, h0 + (size_t)2 * B_ * H_, c0 + (size_t)2 * B_ * H_,
        lengths, nullptr, (float*)d_out);
}
